// SearchLM_10428180594720
// MI455X (gfx1250) — matrix operand order repaired and checked by reading, not compiled, not run on hardware
//
#include <hip/hip_runtime.h>

typedef _Float16 half_t;
typedef __attribute__((ext_vector_type(16))) _Float16 v16h;
typedef __attribute__((ext_vector_type(8)))  _Float16 v8h;
typedef __attribute__((ext_vector_type(8)))  float    v8f;

#define N_HEADS 16
#define D_HEAD  64
#define T_SEQ   1024
#define D_MODEL 1024
#define D_FF    4096
#define N_LAYER 6
#define VOCAB   32000

__device__ __forceinline__ v8f wmma_f16(v16h a, v16h b, v8f c) {
  // D = A(16x32 f16) x B(32x16 f16) + C(16x16 f32)
  return __builtin_amdgcn_wmma_f32_16x16x32_f16(false, a, false, b, (short)0, c,
                                                false, false);
}

// A-operand fragment: lane holds row M=lane&15; K chunks are
// [0..7,16..23] for lanes 0-15 and [8..15,24..31] for lanes 16-31.
__device__ __forceinline__ v16h load_a_frag(const half_t* rowptr, int lane) {
  const int base = (lane & 16) ? 8 : 0;
  v8h lo = *reinterpret_cast<const v8h*>(rowptr + base);
  v8h hi = *reinterpret_cast<const v8h*>(rowptr + base + 16);
  v16h r;
#pragma unroll
  for (int i = 0; i < 8; ++i) { r[i] = lo[i]; r[i + 8] = hi[i]; }
  return r;
}

// B-operand fragment: lane holds column N=lane&15; K chunks are
// [0..7,16..23] for lanes 0-15 and [8..15,24..31] for lanes 16-31.
__device__ __forceinline__ v16h load_b_frag(const half_t* rowptr, int lane) {
  const int base = (lane & 16) ? 8 : 0;
  v8h lo = *reinterpret_cast<const v8h*>(rowptr + base);
  v8h hi = *reinterpret_cast<const v8h*>(rowptr + base + 16);
  v16h r;
#pragma unroll
  for (int i = 0; i < 8; ++i) { r[i] = lo[i]; r[i + 8] = hi[i]; }
  return r;
}

// ---------------- embedding gather ----------------
__global__ void k_embed(const int* __restrict__ x, const float* __restrict__ embed,
                        float* __restrict__ h) {
  const int t = blockIdx.x;
  const long row = (long)x[t] * D_MODEL;
  for (int d = threadIdx.x; d < D_MODEL; d += 256)
    h[(long)t * D_MODEL + d] = embed[row + d];
}

// ---------------- RMSNorm (fp32 in -> f16 out) ----------------
__global__ void k_rmsnorm(const float* __restrict__ h, const float* __restrict__ w,
                          half_t* __restrict__ out, int D) {
  const int t = blockIdx.x;
  const float* row = h + (long)t * D;
  __shared__ float red[256];
  float ss = 0.f;
  for (int d = threadIdx.x; d < D; d += 256) { float v = row[d]; ss += v * v; }
  red[threadIdx.x] = ss;
  __syncthreads();
  for (int s = 128; s > 0; s >>= 1) {
    if (threadIdx.x < s) red[threadIdx.x] += red[threadIdx.x + s];
    __syncthreads();
  }
  const float scale = rsqrtf(red[0] / (float)D + 1e-6f);
  for (int d = threadIdx.x; d < D; d += 256)
    out[(long)t * D + d] = (half_t)(row[d] * scale * w[d]);
}

// ---------------- WMMA GEMM: C[M,N] = A16[M,K] @ Wf32[N,K]^T (+addend) -----
// 256 threads = 8 waves in a 4x2 grid; 128x128 block tile; BK=32.
// A tile staged via CDNA5 async copy (global_load_async_to_lds_b128,
// ASYNCcnt); B tile (fp32 weights) staged through VGPRs with f32->f16
// conversion. Double-buffered LDS: tile k+1 staging overlaps tile-k WMMAs.
__global__ __launch_bounds__(256) void k_gemm(
    const half_t* __restrict__ A, const float* __restrict__ B,
    float* __restrict__ C, const float* __restrict__ addend,
    int M, int N, int K) {
  __shared__ half_t As[2][128][32];
  __shared__ half_t Bs[2][128][32];
  const int tid = threadIdx.x;
  const int lane = tid & 31, wave = tid >> 5;
  const int wy = wave >> 1;        // 0..3 -> 32-row strip
  const int wx = wave & 1;         // 0..1 -> 64-col strip
  const long bm = (long)blockIdx.y * 128;
  const long bn = (long)blockIdx.x * 128;
  const int r = tid >> 1;          // 0..127: staging row
  const int c = (tid & 1) * 16;    // 0 or 16: staging K-chunk (halfs)

  const half_t* aSrc = A + (bm + r) * (long)K + c;
  const float*  bSrc = B + (bn + r) * (long)K + c;

  v8f acc[2][4] = {};
  const int ln = lane & 15;

  // ---- prologue: stage tile 0 into buffer 0 ----
  {
    const unsigned ldsA =
        (unsigned)(unsigned long long)(void*)&As[0][r][c];
    const unsigned long long ga = (unsigned long long)(const void*)aSrc;
    asm volatile("global_load_async_to_lds_b128 %0, %1, off"
                 :: "v"(ldsA), "v"(ga) : "memory");
    asm volatile("global_load_async_to_lds_b128 %0, %1, off offset:16"
                 :: "v"(ldsA), "v"(ga) : "memory");
    const float4 f0 = reinterpret_cast<const float4*>(bSrc)[0];
    const float4 f1 = reinterpret_cast<const float4*>(bSrc)[1];
    const float4 f2 = reinterpret_cast<const float4*>(bSrc)[2];
    const float4 f3 = reinterpret_cast<const float4*>(bSrc)[3];
    v8h h0, h1;
    h0[0] = (half_t)f0.x; h0[1] = (half_t)f0.y; h0[2] = (half_t)f0.z; h0[3] = (half_t)f0.w;
    h0[4] = (half_t)f1.x; h0[5] = (half_t)f1.y; h0[6] = (half_t)f1.z; h0[7] = (half_t)f1.w;
    h1[0] = (half_t)f2.x; h1[1] = (half_t)f2.y; h1[2] = (half_t)f2.z; h1[3] = (half_t)f2.w;
    h1[4] = (half_t)f3.x; h1[5] = (half_t)f3.y; h1[6] = (half_t)f3.z; h1[7] = (half_t)f3.w;
    *reinterpret_cast<v8h*>(&Bs[0][r][c]) = h0;
    *reinterpret_cast<v8h*>(&Bs[0][r][c + 8]) = h1;
  }
  asm volatile("s_wait_asynccnt 0x0" ::: "memory");
  __syncthreads();

  int buf = 0;
  for (int k0 = 32; k0 < K; k0 += 32) {
    // stage next A tile asynchronously into the other buffer
    {
      const unsigned ldsA =
          (unsigned)(unsigned long long)(void*)&As[buf ^ 1][r][c];
      const unsigned long long ga =
          (unsigned long long)(const void*)(aSrc + k0);
      asm volatile("global_load_async_to_lds_b128 %0, %1, off"
                   :: "v"(ldsA), "v"(ga) : "memory");
      asm volatile("global_load_async_to_lds_b128 %0, %1, off offset:16"
                   :: "v"(ldsA), "v"(ga) : "memory");
    }
    // pull next B tile into registers (latency overlapped with WMMAs below)
    const float* src = bSrc + k0;
    const float4 f0 = reinterpret_cast<const float4*>(src)[0];
    const float4 f1 = reinterpret_cast<const float4*>(src)[1];
    const float4 f2 = reinterpret_cast<const float4*>(src)[2];
    const float4 f3 = reinterpret_cast<const float4*>(src)[3];
    if (k0 + 32 < K)
      __builtin_prefetch(bSrc + k0 + 32, 0, 1);

    // compute on current buffer
    {
      v16h a0 = load_a_frag(&As[buf][wy * 32 + ln][0], lane);
      v16h a1 = load_a_frag(&As[buf][wy * 32 + 16 + ln][0], lane);
      v16h b0 = load_b_frag(&Bs[buf][wx * 64 + ln][0], lane);
      v16h b1 = load_b_frag(&Bs[buf][wx * 64 + 16 + ln][0], lane);
      v16h b2 = load_b_frag(&Bs[buf][wx * 64 + 32 + ln][0], lane);
      v16h b3 = load_b_frag(&Bs[buf][wx * 64 + 48 + ln][0], lane);
      acc[0][0] = wmma_f16(a0, b0, acc[0][0]);
      acc[0][1] = wmma_f16(a0, b1, acc[0][1]);
      acc[0][2] = wmma_f16(a0, b2, acc[0][2]);
      acc[0][3] = wmma_f16(a0, b3, acc[0][3]);
      acc[1][0] = wmma_f16(a1, b0, acc[1][0]);
      acc[1][1] = wmma_f16(a1, b1, acc[1][1]);
      acc[1][2] = wmma_f16(a1, b2, acc[1][2]);
      acc[1][3] = wmma_f16(a1, b3, acc[1][3]);
    }
    // convert + store next B tile into the other buffer
    {
      v8h h0, h1;
      h0[0] = (half_t)f0.x; h0[1] = (half_t)f0.y; h0[2] = (half_t)f0.z; h0[3] = (half_t)f0.w;
      h0[4] = (half_t)f1.x; h0[5] = (half_t)f1.y; h0[6] = (half_t)f1.z; h0[7] = (half_t)f1.w;
      h1[0] = (half_t)f2.x; h1[1] = (half_t)f2.y; h1[2] = (half_t)f2.z; h1[3] = (half_t)f2.w;
      h1[4] = (half_t)f3.x; h1[5] = (half_t)f3.y; h1[6] = (half_t)f3.z; h1[7] = (half_t)f3.w;
      *reinterpret_cast<v8h*>(&Bs[buf ^ 1][r][c]) = h0;
      *reinterpret_cast<v8h*>(&Bs[buf ^ 1][r][c + 8]) = h1;
    }
    asm volatile("s_wait_asynccnt 0x0" ::: "memory");
    __syncthreads();
    buf ^= 1;
  }

  // final tile
  {
    v16h a0 = load_a_frag(&As[buf][wy * 32 + ln][0], lane);
    v16h a1 = load_a_frag(&As[buf][wy * 32 + 16 + ln][0], lane);
    v16h b0 = load_b_frag(&Bs[buf][wx * 64 + ln][0], lane);
    v16h b1 = load_b_frag(&Bs[buf][wx * 64 + 16 + ln][0], lane);
    v16h b2 = load_b_frag(&Bs[buf][wx * 64 + 32 + ln][0], lane);
    v16h b3 = load_b_frag(&Bs[buf][wx * 64 + 48 + ln][0], lane);
    acc[0][0] = wmma_f16(a0, b0, acc[0][0]);
    acc[0][1] = wmma_f16(a0, b1, acc[0][1]);
    acc[0][2] = wmma_f16(a0, b2, acc[0][2]);
    acc[0][3] = wmma_f16(a0, b3, acc[0][3]);
    acc[1][0] = wmma_f16(a1, b0, acc[1][0]);
    acc[1][1] = wmma_f16(a1, b1, acc[1][1]);
    acc[1][2] = wmma_f16(a1, b2, acc[1][2]);
    acc[1][3] = wmma_f16(a1, b3, acc[1][3]);
  }

  const int hi8 = (lane >> 4) << 3;
#pragma unroll
  for (int i = 0; i < 2; ++i)
#pragma unroll
    for (int j = 0; j < 4; ++j)
#pragma unroll
      for (int rr = 0; rr < 8; ++rr) {
        const long m = bm + wy * 32 + i * 16 + rr + hi8;
        const long n = bn + wx * 64 + j * 16 + ln;
        float v = acc[i][j][rr];
        if (addend) v += addend[m * (long)N + n];
        C[m * (long)N + n] = v;
      }
}

// ---------------- per-head RMS + RoPE on q,k; transpose-convert v ---------
__global__ void k_qkv_prep(const float* __restrict__ q32, const float* __restrict__ k32,
                           const float* __restrict__ v32,
                           const float* __restrict__ qn, const float* __restrict__ kn,
                           half_t* __restrict__ q16, half_t* __restrict__ k16,
                           half_t* __restrict__ v16t) {
  const int t = blockIdx.x, hh = blockIdx.y, d = threadIdx.x;
  __shared__ float sq[D_HEAD], sk[D_HEAD], red[D_HEAD];
  const long idx = (long)t * D_MODEL + hh * D_HEAD + d;
  const float qv = q32[idx], kv = k32[idx], vv = v32[idx];

  red[d] = qv * qv;
  __syncthreads();
  for (int s = 32; s > 0; s >>= 1) { if (d < s) red[d] += red[d + s]; __syncthreads(); }
  const float qs = rsqrtf(red[0] / (float)D_HEAD + 1e-6f);
  __syncthreads();
  red[d] = kv * kv;
  __syncthreads();
  for (int s = 32; s > 0; s >>= 1) { if (d < s) red[d] += red[d + s]; __syncthreads(); }
  const float ks = rsqrtf(red[0] / (float)D_HEAD + 1e-6f);
  __syncthreads();

  sq[d] = qv * qs * qn[d];
  sk[d] = kv * ks * kn[d];
  __syncthreads();

  if (d < 32) {
    const int i = d;
    const float inv_freq = __powf(10000.0f, -(2.0f * i) / (float)D_HEAD);
    const float ang = (float)t * inv_freq;
    const float cc = __cosf(ang), ss = __sinf(ang);
    const long base = ((long)hh * T_SEQ + t) * D_HEAD;
    float x1 = sq[2 * i], x2 = sq[2 * i + 1];
    q16[base + 2 * i]     = (half_t)(x1 * cc - x2 * ss);
    q16[base + 2 * i + 1] = (half_t)(x1 * ss + x2 * cc);
    x1 = sk[2 * i]; x2 = sk[2 * i + 1];
    k16[base + 2 * i]     = (half_t)(x1 * cc - x2 * ss);
    k16[base + 2 * i + 1] = (half_t)(x1 * ss + x2 * cc);
  }
  v16t[((long)hh * D_HEAD + d) * T_SEQ + t] = (half_t)vv;
}

// ---------------- flash attention: one wave per (16-query tile, head) -----
__global__ __launch_bounds__(32) void k_attn(
    const half_t* __restrict__ q16, const half_t* __restrict__ k16,
    const half_t* __restrict__ v16t, half_t* __restrict__ o16) {
  const int qt = blockIdx.x, hh = blockIdx.y;
  const int lane = threadIdx.x & 31;
  const int ln = lane & 15;
  const int hi8 = (lane >> 4) << 3;
  __shared__ half_t pbuf[16][32];   // P-tile bounce: C-layout -> A-layout

  const half_t* qbase = q16 + ((long)hh * T_SEQ + qt * 16) * D_HEAD;
  const v16h qa0 = load_a_frag(qbase + (long)ln * D_HEAD, lane);       // dh 0..31
  const v16h qa1 = load_a_frag(qbase + (long)ln * D_HEAD + 32, lane);  // dh 32..63

  v8f o[4] = {};
  float m_i[8], l_i[8];
#pragma unroll
  for (int rr = 0; rr < 8; ++rr) { m_i[rr] = -1e30f; l_i[rr] = 0.f; }

  const half_t* kb_h = k16 + (long)hh * T_SEQ * D_HEAD;
  const half_t* vb_h = v16t + (long)hh * D_HEAD * T_SEQ;
  const int kend = qt * 16 + 16;
  const int steps = (kend + 31) / 32;

  for (int s32 = 0; s32 < steps; ++s32) {
    const int kbase = s32 * 32;
    v8f s0 = {}, s1 = {};
    {
      const half_t* krow0 = kb_h + (long)(kbase + ln) * D_HEAD;
      s0 = wmma_f16(qa0, load_b_frag(krow0, lane), s0);
      s0 = wmma_f16(qa1, load_b_frag(krow0 + 32, lane), s0);
      const half_t* krow1 = kb_h + (long)(kbase + 16 + ln) * D_HEAD;
      s1 = wmma_f16(qa0, load_b_frag(krow1, lane), s1);
      s1 = wmma_f16(qa1, load_b_frag(krow1 + 32, lane), s1);
    }
    float corr[8];
#pragma unroll
    for (int rr = 0; rr < 8; ++rr) {
      const int qrow = qt * 16 + rr + hi8;
      float e0 = 50.0f * tanhf(s0[rr] * 0.125f * (1.0f / 50.0f));
      float e1 = 50.0f * tanhf(s1[rr] * 0.125f * (1.0f / 50.0f));
      if (kbase + ln > qrow) e0 = -1e30f;          // causal mask (after softcap)
      if (kbase + 16 + ln > qrow) e1 = -1e30f;
      float mx = fmaxf(e0, e1);
#pragma unroll
      for (int m = 1; m < 16; m <<= 1) mx = fmaxf(mx, __shfl_xor(mx, m, 32));
      const float mnew = fmaxf(m_i[rr], mx);
      const float cf = __expf(m_i[rr] - mnew);
      const float p0 = __expf(e0 - mnew), p1 = __expf(e1 - mnew);
      float rs = p0 + p1;
#pragma unroll
      for (int m = 1; m < 16; m <<= 1) rs += __shfl_xor(rs, m, 32);
      l_i[rr] = l_i[rr] * cf + rs;
      m_i[rr] = mnew;
      corr[rr] = cf;
      pbuf[rr + hi8][ln] = (half_t)p0;
      pbuf[rr + hi8][16 + ln] = (half_t)p1;
    }
#pragma unroll
    for (int j = 0; j < 4; ++j)
#pragma unroll
      for (int rr = 0; rr < 8; ++rr) o[j][rr] *= corr[rr];
    // single-wave LDS RAW: wait for the ds stores before the transposed read
    asm volatile("s_wait_dscnt 0x0" ::: "memory");
    const v16h pa = load_a_frag(&pbuf[ln][0], lane);
#pragma unroll
    for (int j = 0; j < 4; ++j) {
      const half_t* vrow = vb_h + (long)(j * 16 + ln) * T_SEQ + kbase;
      o[j] = wmma_f16(pa, load_b_frag(vrow, lane), o[j]);
    }
  }

#pragma unroll
  for (int j = 0; j < 4; ++j)
#pragma unroll
    for (int rr = 0; rr < 8; ++rr) {
      const int trow = qt * 16 + rr + hi8;
      o16[(long)trow * D_MODEL + hh * D_HEAD + j * 16 + ln] =
          (half_t)(o[j][rr] / l_i[rr]);
    }
}

// ---------------- SwiGLU fuse: m16 = f16(silu(g) * u) ----------------
__global__ void k_swiglu(const float* __restrict__ g, const float* __restrict__ u,
                         half_t* __restrict__ out, long n) {
  const long i = (long)blockIdx.x * blockDim.x + threadIdx.x;
  if (i < n) {
    const float gv = g[i];
    const float sg = gv / (1.0f + __expf(-gv));
    out[i] = (half_t)(sg * u[i]);
  }
}

extern "C" void kernel_launch(void* const* d_in, const int* in_sizes, int n_in,
                              void* d_out, int out_size, void* d_ws, size_t ws_size,
                              hipStream_t stream) {
  (void)in_sizes; (void)n_in; (void)out_size; (void)ws_size;
  const int*   x      = (const int*)d_in[0];
  const float* embed  = (const float*)d_in[1];
  const float* ln1    = (const float*)d_in[2];
  const float* Wq     = (const float*)d_in[3];
  const float* Wk     = (const float*)d_in[4];
  const float* Wv     = (const float*)d_in[5];
  const float* Wo     = (const float*)d_in[6];
  const float* qn     = (const float*)d_in[7];
  const float* kn     = (const float*)d_in[8];
  const float* ln2    = (const float*)d_in[9];
  const float* Wg     = (const float*)d_in[10];
  const float* Wu     = (const float*)d_in[11];
  const float* Wd     = (const float*)d_in[12];
  const float* ln_out = (const float*)d_in[13];
  float* logits = (float*)d_out;

  char* p = (char*)d_ws;
  auto alloc = [&](size_t bytes) { char* r = p; p += (bytes + 255) & ~(size_t)255; return r; };
  float*  h    = (float*)alloc((size_t)T_SEQ * D_MODEL * 4);
  half_t* a16  = (half_t*)alloc((size_t)T_SEQ * D_MODEL * 2);
  float*  q32  = (float*)alloc((size_t)T_SEQ * D_MODEL * 4);
  float*  k32  = (float*)alloc((size_t)T_SEQ * D_MODEL * 4);
  float*  v32  = (float*)alloc((size_t)T_SEQ * D_MODEL * 4);
  half_t* q16  = (half_t*)alloc((size_t)T_SEQ * D_MODEL * 2);
  half_t* k16  = (half_t*)alloc((size_t)T_SEQ * D_MODEL * 2);
  half_t* v16t = (half_t*)alloc((size_t)T_SEQ * D_MODEL * 2);
  half_t* o16  = (half_t*)alloc((size_t)T_SEQ * D_MODEL * 2);
  float*  g32  = (float*)alloc((size_t)T_SEQ * D_FF * 4);
  float*  u32  = (float*)alloc((size_t)T_SEQ * D_FF * 4);
  half_t* m16  = (half_t*)alloc((size_t)T_SEQ * D_FF * 2);

  const dim3 gemmBlk(256);
  const dim3 gD(D_MODEL / 128, T_SEQ / 128);     // 8 x 8
  const dim3 gFF(D_FF / 128, T_SEQ / 128);       // 32 x 8
  const dim3 gV(VOCAB / 128, T_SEQ / 128);       // 250 x 8

  k_embed<<<T_SEQ, 256, 0, stream>>>(x, embed, h);

  for (int l = 0; l < N_LAYER; ++l) {
    const long wOff  = (long)l * D_MODEL * D_MODEL;
    const long ffOff = (long)l * (long)D_FF * D_MODEL;

    k_rmsnorm<<<T_SEQ, 256, 0, stream>>>(h, ln1 + (long)l * D_MODEL, a16, D_MODEL);
    k_gemm<<<gD, gemmBlk, 0, stream>>>(a16, Wq + wOff, q32, nullptr, T_SEQ, D_MODEL, D_MODEL);
    k_gemm<<<gD, gemmBlk, 0, stream>>>(a16, Wk + wOff, k32, nullptr, T_SEQ, D_MODEL, D_MODEL);
    k_gemm<<<gD, gemmBlk, 0, stream>>>(a16, Wv + wOff, v32, nullptr, T_SEQ, D_MODEL, D_MODEL);

    k_qkv_prep<<<dim3(T_SEQ, N_HEADS), 64, 0, stream>>>(
        q32, k32, v32, qn + (long)l * D_HEAD, kn + (long)l * D_HEAD, q16, k16, v16t);

    k_attn<<<dim3(T_SEQ / 16, N_HEADS), 32, 0, stream>>>(q16, k16, v16t, o16);

    k_gemm<<<gD, gemmBlk, 0, stream>>>(o16, Wo + wOff, h, h, T_SEQ, D_MODEL, D_MODEL);

    k_rmsnorm<<<T_SEQ, 256, 0, stream>>>(h, ln2 + (long)l * D_MODEL, a16, D_MODEL);
    k_gemm<<<gFF, gemmBlk, 0, stream>>>(a16, Wg + ffOff, g32, nullptr, T_SEQ, D_FF, D_MODEL);
    k_gemm<<<gFF, gemmBlk, 0, stream>>>(a16, Wu + ffOff, u32, nullptr, T_SEQ, D_FF, D_MODEL);

    const long nFF = (long)T_SEQ * D_FF;
    k_swiglu<<<(int)((nFF + 255) / 256), 256, 0, stream>>>(g32, u32, m16, nFF);

    k_gemm<<<gD, gemmBlk, 0, stream>>>(m16, Wd + ffOff, h, h, T_SEQ, D_MODEL, D_FF);
  }

  k_rmsnorm<<<T_SEQ, 256, 0, stream>>>(h, ln_out, a16, D_MODEL);
  k_gemm<<<gV, gemmBlk, 0, stream>>>(a16, embed, logits, nullptr, T_SEQ, VOCAB, D_MODEL);
}
